// SNN_43482248904985
// MI455X (gfx1250) — 8-bit matrix operand order repaired and checked by reading, compiled, not run on hardware
//
#include <hip/hip_runtime.h>


typedef __attribute__((ext_vector_type(8))) int v8i;

#define N_STEPS 1000
#define T_PAD   1008
#define NUM_IN  700
#define NI_PAD  704
#define NH      4096
#define NO      20
#define NO_PAD  32
#define NWG     64
#define TPB     128

constexpr size_t OFF_SPK8  = 0;
constexpr size_t OFF_W1    = OFF_SPK8  + (size_t)T_PAD * NI_PAD;
constexpr size_t OFF_V1    = OFF_W1    + (size_t)NH * NI_PAD;
constexpr size_t OFF_W2T   = OFF_V1    + (size_t)NH * NH;
constexpr size_t OFF_INALL = OFF_W2T   + (size_t)NH * NO_PAD;
constexpr size_t OFF_SBUF  = OFF_INALL + (size_t)T_PAD * NH * 4;
constexpr size_t OFF_BAR   = OFF_SBUF  + (size_t)2 * NH;

__global__ void snn_pack_kernel(const int* __restrict__ spk,
                                const int* __restrict__ w1,
                                const int* __restrict__ v1,
                                const int* __restrict__ w2,
                                unsigned char* __restrict__ ws) {
  size_t i = (size_t)blockIdx.x * blockDim.x + threadIdx.x;
  size_t stride = (size_t)gridDim.x * blockDim.x;

  char* spk8 = (char*)(ws + OFF_SPK8);
  char* w18  = (char*)(ws + OFF_W1);
  char* v18  = (char*)(ws + OFF_V1);
  char* w2t  = (char*)(ws + OFF_W2T);
  char* sbuf = (char*)(ws + OFF_SBUF);
  int*  bar  = (int*)(ws + OFF_BAR);

  for (size_t j = i; j < (size_t)T_PAD * NI_PAD; j += stride) {
    int t = (int)(j / NI_PAD), k = (int)(j % NI_PAD);
    spk8[j] = (t < N_STEPS && k < NUM_IN) ? (char)spk[(size_t)t * NUM_IN + k] : 0;
  }
  for (size_t j = i; j < (size_t)NH * NI_PAD; j += stride) {
    int h = (int)(j / NI_PAD), k = (int)(j % NI_PAD);
    w18[j] = (k < NUM_IN) ? (char)w1[(size_t)h * NUM_IN + k] : 0;
  }
  for (size_t j = i; j < (size_t)NH * NH; j += stride) {
    v18[j] = (char)v1[j];
  }
  for (size_t j = i; j < (size_t)NH * NO_PAD; j += stride) {
    int h = (int)(j / NO_PAD), o = (int)(j % NO_PAD);
    w2t[j] = (o < NO) ? (char)w2[(size_t)o * NH + h] : 0;
  }
  for (size_t j = i; j < (size_t)2 * NH; j += stride) sbuf[j] = 0;
  if (i == 0) { bar[0] = 0; bar[1] = 0; }
}

__global__ __launch_bounds__(256) void snn_in_gemm_kernel(const unsigned char* __restrict__ ws_in,
                                                          unsigned char* __restrict__ ws) {
  const char* spk8 = (const char*)(ws_in + OFF_SPK8);
  const char* w18  = (const char*)(ws_in + OFF_W1);
  int* in_all = (int*)(ws + OFF_INALL);

  int lane = threadIdx.x & 31;
  int wid  = blockIdx.x * 8 + (threadIdx.x >> 5);
  int tileT = wid >> 8;
  int tileH = wid & 255;
  int m  = lane & 15;
  int hi = lane >> 4;

  v8i acc = {0, 0, 0, 0, 0, 0, 0, 0};
  size_t arow = (size_t)(tileT * 16 + m) * NI_PAD;
  size_t brow = (size_t)(tileH * 16 + m) * NI_PAD;

  for (int kc = 0; kc < NI_PAD / 64; ++kc) {
    int kbase = kc * 64;
    v8i a;
    size_t abase = arow + kbase + hi * 8;
#pragma unroll
    for (int p = 0; p < 4; ++p) {
      uint2 q = *(const uint2*)(spk8 + abase + p * 16);
      a[2 * p] = (int)q.x; a[2 * p + 1] = (int)q.y;
    }
    size_t bbase = brow + kbase + hi * 8;
    v8i b;
#pragma unroll
    for (int p = 0; p < 4; ++p) {
      uint2 q = *(const uint2*)(w18 + bbase + p * 16);
      b[2 * p] = (int)q.x; b[2 * p + 1] = (int)q.y;
    }
    acc = __builtin_amdgcn_wmma_i32_16x16x64_iu8(false, a, true, b, acc, false, false);
  }
  int hcol = tileH * 16 + m;
  int trow = tileT * 16 + hi * 8;
#pragma unroll
  for (int r = 0; r < 8; ++r)
    in_all[(size_t)(trow + r) * NH + hcol] = acc[r];
}

__device__ inline void snn_grid_barrier(int* bar) {
  __syncthreads();
  if (threadIdx.x == 0) {
    __threadfence();
    int g = __hip_atomic_load(&bar[1], __ATOMIC_ACQUIRE, __HIP_MEMORY_SCOPE_AGENT);
    if (__hip_atomic_fetch_add(&bar[0], 1, __ATOMIC_ACQ_REL, __HIP_MEMORY_SCOPE_AGENT) == NWG - 1) {
      __hip_atomic_store(&bar[0], 0, __ATOMIC_RELAXED, __HIP_MEMORY_SCOPE_AGENT);
      __hip_atomic_fetch_add(&bar[1], 1, __ATOMIC_ACQ_REL, __HIP_MEMORY_SCOPE_AGENT);
    } else {
      while (__hip_atomic_load(&bar[1], __ATOMIC_ACQUIRE, __HIP_MEMORY_SCOPE_AGENT) == g)
        __builtin_amdgcn_s_sleep(1);
    }
  }
  __syncthreads();
  __threadfence();
}

__global__ __launch_bounds__(TPB) void snn_recurrent_kernel(unsigned char* __restrict__ ws,
                                                            float* __restrict__ out) {
  const char* v18    = (const char*)(ws + OFF_V1);
  const int*  in_all = (const int*)(ws + OFF_INALL);
  const char* w2t    = (const char*)(ws + OFF_W2T);
  char* sbuf0 = (char*)(ws + OFF_SBUF);
  char* sbuf1 = sbuf0 + NH;
  int*  bar   = (int*)(ws + OFF_BAR);

  __shared__ int lds_fb[64];
  __shared__ int lds_ro[NO];

  const int tid  = threadIdx.x;
  const int lane = tid & 31;
  const int wave = tid >> 5;
  const int m    = lane & 15;
  const int hi   = lane >> 4;
  const int wg   = blockIdx.x;
  const size_t arow = (size_t)(wg * 64 + wave * 16 + m) * NH;
  const int neuron = wg * 64 + tid;

  int mem = 0, syn = 0, prev_out = 0;
  int rmem = 0, rsyn = 0;

  for (int t = 0; t < N_STEPS; ++t) {
    const char* sin = (t & 1) ? sbuf1 : sbuf0;
    char* sout      = (t & 1) ? sbuf0 : sbuf1;

    v8i acc = {0, 0, 0, 0, 0, 0, 0, 0};
    for (int kc = 0; kc < NH / 64; ++kc) {
      int kbase = kc * 64;
      v8i a;
      size_t abase = arow + kbase + hi * 8;
#pragma unroll
      for (int p = 0; p < 4; ++p) {
        uint2 q = *(const uint2*)(v18 + abase + p * 16);
        a[2 * p] = (int)q.x; a[2 * p + 1] = (int)q.y;
      }
      v8i b = {0, 0, 0, 0, 0, 0, 0, 0};
      if (m == 0) {
#pragma unroll
        for (int p = 0; p < 4; ++p) {
          uint2 q = *(const uint2*)(sin + kbase + hi * 8 + p * 16);
          b[2 * p] = (int)q.x; b[2 * p + 1] = (int)q.y;
        }
      }
      acc = __builtin_amdgcn_wmma_i32_16x16x64_iu8(true, a, false, b, acc, false, false);
    }
    if (m == 0) {
#pragma unroll
      for (int r = 0; r < 8; ++r) lds_fb[wave * 16 + hi * 8 + r] = acc[r];
    }
    __syncthreads();

    if (tid < 64) {
      int fb = lds_fb[tid];
      int syn_curr = in_all[(size_t)t * NH + neuron] + fb;
      int mr   = prev_out ? 0 : mem;
      int nout = (mr - 1) > 0 ? 1 : 0;
      mem = mr - (mr >> 3) + syn;
      syn = syn - (syn >> 4) + syn_curr;
      prev_out = nout;
      sout[neuron] = (char)nout;
    }

    snn_grid_barrier(bar);

    if (wg == 0) {
      if (tid < NO) lds_ro[tid] = 0;
      __syncthreads();
      int acc_o[NO];
#pragma unroll
      for (int o = 0; o < NO; ++o) acc_o[o] = 0;
      for (int h = tid; h < NH; h += TPB) {
        if (sout[h]) {
          const int* wr = (const int*)(w2t + (size_t)h * NO_PAD);
#pragma unroll
          for (int w = 0; w < 5; ++w) {
            int pk = wr[w];
#pragma unroll
            for (int b = 0; b < 4; ++b)
              acc_o[w * 4 + b] += (pk << (24 - 8 * b)) >> 24;
          }
        }
      }
#pragma unroll
      for (int o = 0; o < NO; ++o) atomicAdd(&lds_ro[o], acc_o[o]);
      __syncthreads();
      if (tid < NO) {
        int ro   = lds_ro[tid];
        int nrm  = rmem - (rmem >> 3) + rsyn;
        rsyn = rsyn - (rsyn >> 4) + ro;
        rmem = nrm;
        out[tid * N_STEPS + t] = (float)nrm;
      }
      __syncthreads();
    }
  }
}

extern "C" void kernel_launch(void* const* d_in, const int* in_sizes, int n_in,
                              void* d_out, int out_size, void* d_ws, size_t ws_size,
                              hipStream_t stream) {
  (void)in_sizes; (void)n_in; (void)out_size; (void)ws_size;
  const int* spk = (const int*)d_in[0];
  const int* w1  = (const int*)d_in[1];
  const int* v1  = (const int*)d_in[2];
  const int* w2  = (const int*)d_in[3];
  unsigned char* ws = (unsigned char*)d_ws;
  float* out = (float*)d_out;

  snn_pack_kernel<<<4096, 256, 0, stream>>>(spk, w1, v1, w2, ws);
  snn_in_gemm_kernel<<<2016, 256, 0, stream>>>(ws, ws);
  snn_recurrent_kernel<<<NWG, TPB, 0, stream>>>(ws, out);
}
